// CTLSTMCell_21973052686346
// MI455X (gfx1250) — matrix operand order repaired and checked by reading, not compiled, not run on hardware
//
#include <hip/hip_runtime.h>

#define D_UNITS 1024
#define BATCH_N 4096
#define KDIM    2048   // 2*D
#define NGATE   7
#define NCOL    (NGATE * D_UNITS)   // 7168
#define BM      128
#define BN      32
#define KT      32
#define LDSR    40     // padded row stride (elements) => 80B, conflict-free
#define NT      (KDIM / KT)         // 64

typedef __attribute__((ext_vector_type(8)))  float  v8f;
typedef __attribute__((ext_vector_type(8)))  __bf16 v8bf;
typedef __attribute__((ext_vector_type(16))) __bf16 v16bf;
typedef __attribute__((ext_vector_type(4)))  int    v4i;

// pointer-to-int4 in global (AS1) / LDS (AS3) address spaces
typedef __attribute__((address_space(1))) v4i* as1_v4i_p;
typedef __attribute__((address_space(3))) v4i* as3_v4i_p;

#if defined(__HIP_DEVICE_COMPILE__)
#if __has_builtin(__builtin_amdgcn_global_load_async_to_lds_b128)
#define HAVE_ASYNC_LDS 1
#else
#define HAVE_ASYNC_LDS 0
#endif
#if __has_builtin(__builtin_amdgcn_sched_barrier)
#define SCHED_FENCE() __builtin_amdgcn_sched_barrier(0)
#else
#define SCHED_FENCE()
#endif
#else
#define HAVE_ASYNC_LDS 0
#define SCHED_FENCE()
#endif

// ---------------- prep kernels: f32 -> bf16 (A concat, W transposed) --------

__global__ void ctlstm_prep_a(const float* __restrict__ emb,
                              const float* __restrict__ hid,
                              unsigned short* __restrict__ abf_raw) {
  __bf16* abf = reinterpret_cast<__bf16*>(abf_raw);
  size_t i = (size_t)blockIdx.x * blockDim.x + threadIdx.x;
  int row = (int)(i >> 11);        // / KDIM
  int k   = (int)(i & (KDIM - 1));
  float v = (k < D_UNITS) ? emb[(size_t)row * D_UNITS + k]
                          : hid[(size_t)row * D_UNITS + (k - D_UNITS)];
  abf[i] = (__bf16)v;
}

// Wt[n][k] = (bf16) W[k][n] : N-major so B fragments read contiguous K
__global__ void ctlstm_prep_w(const float* __restrict__ W,
                              unsigned short* __restrict__ wt_raw) {
  __bf16* wt = reinterpret_cast<__bf16*>(wt_raw);
  __shared__ float tile[32][33];
  const int k0 = blockIdx.x * 32;
  const int n0 = blockIdx.y * 32;
  const int tx = threadIdx.x;      // 0..31
  const int ty = threadIdx.y;      // 0..7
#pragma unroll
  for (int i = 0; i < 4; ++i)
    tile[ty + 8 * i][tx] = W[(size_t)(k0 + ty + 8 * i) * NCOL + n0 + tx];
  __syncthreads();
#pragma unroll
  for (int i = 0; i < 4; ++i)
    wt[(size_t)(n0 + ty + 8 * i) * KDIM + k0 + tx] = (__bf16)tile[tx][ty + 8 * i];
}

// ---------------- fused GEMM + gate activations -----------------------------

__device__ __forceinline__ float sigmoid_f(float x) {
  return 1.0f / (1.0f + __expf(-x));
}
__device__ __forceinline__ float tanh_f(float x) {
  float e = __expf(2.0f * x);
  return 1.0f - 2.0f / (e + 1.0f);   // safe for +/- inf
}

#if HAVE_ASYNC_LDS
__device__ __forceinline__
void async_cp16(const __bf16* g, __bf16* l) {
  // global -> AS(1) int4* via integer cast; LDS generic low 32 bits are the
  // LDS byte address (ISA aperture rule), so truncate for the AS(3) pointer.
  as1_v4i_p gp = (as1_v4i_p)(unsigned long long)(__UINTPTR_TYPE__)g;
  as3_v4i_p lp = (as3_v4i_p)(unsigned int)(__UINTPTR_TYPE__)l;
  __builtin_amdgcn_global_load_async_to_lds_b128(gp, lp, 0, 0);
}
__device__ __forceinline__ void async_wait0() {
#if __has_builtin(__builtin_amdgcn_s_wait_asynccnt)
  __builtin_amdgcn_s_wait_asynccnt(0);
#else
  asm volatile("s_wait_asynccnt 0x0" ::: "memory");
#endif
}
#endif

__device__ __forceinline__
void stage_tiles(const __bf16* __restrict__ Ag, const __bf16* __restrict__ Wg,
                 __bf16* As, __bf16* Ws, int m0, int d0, int k0, int tid) {
  // A tile: 128 rows x 32 k -> 512 x 16B chunks; W tile: 224 rows -> 896 chunks
#pragma unroll
  for (int i = 0; i < 2; ++i) {
    int c = tid + i * 256;
    int row = c >> 2, sect = c & 3;
    const __bf16* g = Ag + (size_t)(m0 + row) * KDIM + k0 + sect * 8;
    __bf16* l = As + row * LDSR + sect * 8;
#if HAVE_ASYNC_LDS
    async_cp16(g, l);
#else
    *reinterpret_cast<uint4*>(l) = *reinterpret_cast<const uint4*>(g);
#endif
  }
#pragma unroll
  for (int i = 0; i < 4; ++i) {
    int c = tid + i * 256;
    if (c < NGATE * BN * 4) {
      int row = c >> 2, sect = c & 3;
      int gi = row >> 5, nl = row & 31;
      const __bf16* g = Wg + (size_t)(gi * D_UNITS + d0 + nl) * KDIM + k0 + sect * 8;
      __bf16* l = Ws + row * LDSR + sect * 8;
#if HAVE_ASYNC_LDS
      async_cp16(g, l);
#else
      *reinterpret_cast<uint4*>(l) = *reinterpret_cast<const uint4*>(g);
#endif
    }
  }
}

__device__ __forceinline__
v16bf load_frag(const __bf16* p0, const __bf16* p1) {
  v8bf lo = *reinterpret_cast<const v8bf*>(p0);
  v8bf hi = *reinterpret_cast<const v8bf*>(p1);
  return __builtin_shufflevector(lo, hi, 0, 1, 2, 3, 4, 5, 6, 7,
                                 8, 9, 10, 11, 12, 13, 14, 15);
}

__global__ __launch_bounds__(256)
void ctlstm_gemm_fused(const unsigned short* __restrict__ abf_raw,
                       const unsigned short* __restrict__ wt_raw,
                       const float* __restrict__ bias,
                       const float* __restrict__ cprev,
                       const float* __restrict__ cbprev,
                       float* __restrict__ out) {
  const __bf16* Ag = reinterpret_cast<const __bf16*>(abf_raw);
  const __bf16* Wg = reinterpret_cast<const __bf16*>(wt_raw);

  __shared__ alignas(16) __bf16 Ash[2][BM * LDSR];           // 2 x 10KB
  __shared__ alignas(16) __bf16 Wsh[2][NGATE * BN * LDSR];   // 2 x 17.5KB

  const int tid  = threadIdx.x;
  const int wave = tid >> 5;
  const int lane = tid & 31;
  const int half = lane >> 4;
  const int l16  = lane & 15;

  const int m0 = (int)(blockIdx.x >> 5) * BM;   // 32 row tiles
  const int d0 = (int)(blockIdx.x & 31) * BN;   // 32 col tiles within D

  v8f acc[NGATE][2];
#pragma unroll
  for (int g = 0; g < NGATE; ++g)
#pragma unroll
    for (int j = 0; j < 2; ++j) acc[g][j] = (v8f)0.0f;

  stage_tiles(Ag, Wg, Ash[0], Wsh[0], m0, d0, 0, tid);
#if HAVE_ASYNC_LDS
  async_wait0();
#endif
  __syncthreads();

  for (int kt = 0; kt < NT; ++kt) {
    const int cur = kt & 1;
    if (kt + 1 < NT)
      stage_tiles(Ag, Wg, Ash[cur ^ 1], Wsh[cur ^ 1], m0, d0, (kt + 1) * KT, tid);
    if (kt + 2 < NT) {
      // L2 prefetch of the k+2 tiles (global_prefetch_b8)
      int row = tid >> 2, sect = tid & 3;
      __builtin_prefetch(Ag + (size_t)(m0 + row) * KDIM + (kt + 2) * KT + sect * 8, 0, 1);
      int g = row >> 5, nl = row & 31;
      __builtin_prefetch(Wg + (size_t)(g * D_UNITS + d0 + nl) * KDIM + (kt + 2) * KT + sect * 8, 0, 1);
    }

    // A fragment (16x32 bf16): lanes 0-15 -> K{0-7,16-23}, lanes 16-31 -> K{8-15,24-31}
    const __bf16* Ab = Ash[cur] + (wave * 16 + l16) * LDSR;
    v16bf afrag = load_frag(Ab + half * 8, Ab + 16 + half * 8);

    // B fragments (32x16): lane holds column N=l16; lanes 0-15 K{0-7,16-23}, 16-31 K{8-15,24-31}.
    // Load 7 fragments as a group, fence the scheduler, then 7 back-to-back WMMAs.
#pragma unroll
    for (int j = 0; j < 2; ++j) {
      v16bf bf[NGATE];
#pragma unroll
      for (int g = 0; g < NGATE; ++g) {
        const __bf16* Bb = Wsh[cur] + (g * BN + j * 16 + l16) * LDSR + half * 8;
        bf[g] = load_frag(Bb, Bb + 16);
      }
      SCHED_FENCE();
#pragma unroll
      for (int g = 0; g < NGATE; ++g) {
        acc[g][j] = __builtin_amdgcn_wmma_f32_16x16x32_bf16(
            false, afrag, false, bf[g], (short)0, acc[g][j], false, false);
      }
      SCHED_FENCE();
    }

#if HAVE_ASYNC_LDS
    async_wait0();   // stage(kt+1) copies had all of compute(kt) to finish
#endif
    __syncthreads();
  }

  // -------- fused epilogue: biases + activations + cell updates -------------
  const size_t BD = (size_t)BATCH_N * D_UNITS;
  float* out_cell  = out;
  float* out_cbar  = out + BD;
  float* out_decay = out + 2 * BD;
  float* out_og    = out + 3 * BD;

  // C layout: VGPR r -> M = r + 8*half (within wave's 16 rows), N = l16
  const int mbase = m0 + wave * 16 + 8 * half;

#pragma unroll
  for (int j = 0; j < 2; ++j) {
    const int n = d0 + j * 16 + l16;
    float bg[NGATE];
#pragma unroll
    for (int g = 0; g < NGATE; ++g) bg[g] = bias[g * D_UNITS + n];
#pragma unroll
    for (int r = 0; r < 8; ++r) {
      const size_t idx = (size_t)(mbase + r) * D_UNITS + n;
      const float cp  = cprev[idx];
      const float cbp = cbprev[idx];
      const float ig = sigmoid_f(acc[0][j][r] + bg[0]);
      const float fg = sigmoid_f(acc[1][j][r] + bg[1]);
      const float og = sigmoid_f(acc[2][j][r] + bg[2]);
      const float z  = tanh_f(acc[3][j][r] + bg[3]);
      const float ib = sigmoid_f(acc[4][j][r] + bg[4]);
      const float fb = sigmoid_f(acc[5][j][r] + bg[5]);
      const float dz = 0.1f * (acc[6][j][r] + bg[6]);
      // softplus(dz)/0.1, numerically stable
      const float dec = 10.0f * (fmaxf(dz, 0.0f) +
                                 __logf(1.0f + __expf(-fabsf(dz))));
      out_cell[idx]  = fg * cp + ig * z;
      out_cbar[idx]  = fb * cbp + ib * z;
      out_decay[idx] = dec;
      out_og[idx]    = og;
    }
  }
}

// ---------------- launch ----------------------------------------------------

extern "C" void kernel_launch(void* const* d_in, const int* in_sizes, int n_in,
                              void* d_out, int out_size, void* d_ws, size_t ws_size,
                              hipStream_t stream) {
  const float* emb    = (const float*)d_in[0];
  const float* hid    = (const float*)d_in[1];
  const float* cprev  = (const float*)d_in[2];
  const float* cbprev = (const float*)d_in[3];
  const float* W      = (const float*)d_in[4];
  const float* b      = (const float*)d_in[5];
  float* out          = (float*)d_out;

  // workspace: A as bf16 [4096,2048] (16MB) then W^T as bf16 [7168,2048] (28MB)
  unsigned short* abf = (unsigned short*)d_ws;
  unsigned short* wtb = abf + (size_t)BATCH_N * KDIM;

  ctlstm_prep_a<<<(BATCH_N * KDIM) / 256, 256, 0, stream>>>(emb, hid, abf);
  ctlstm_prep_w<<<dim3(KDIM / 32, NCOL / 32), dim3(32, 8), 0, stream>>>(W, wtb);
  ctlstm_gemm_fused<<<(BATCH_N / BM) * (D_UNITS / BN), 256, 0, stream>>>(
      abf, wtb, b, cprev, cbprev, out);
}
